// TransformerBlock_13434657701969
// MI455X (gfx1250) — matrix operand order repaired and checked by reading, not compiled, not run on hardware
//
#include <hip/hip_runtime.h>

// ---------------------------------------------------------------------------
// Problem constants (from reference)
// ---------------------------------------------------------------------------
#define B_   8
#define C_   192
#define H_   128
#define W_   128
#define HW_  16384            // H_*W_
#define HEADS_ 4
#define CH_  48               // C_/HEADS_
#define KIN_ 192              // K dim of every 1x1 conv

typedef __attribute__((ext_vector_type(16))) __bf16 v16bf;
typedef __attribute__((ext_vector_type(8)))  float  v8f;
typedef unsigned int v4u __attribute__((ext_vector_type(4)));
typedef unsigned int v8u __attribute__((ext_vector_type(8)));

#if defined(__AMDGCN__)
#define WAIT_TENSOR() __builtin_amdgcn_s_wait_tensorcnt(0)
#else
#define WAIT_TENSOR() ((void)0)
#endif

// ---------------------------------------------------------------------------
// bf16 helpers (stored as raw ushort bits in workspace buffers)
// ---------------------------------------------------------------------------
__device__ __forceinline__ float bf2f(unsigned short u) {
    union { unsigned int i; float f; } t; t.i = ((unsigned int)u) << 16; return t.f;
}
__device__ __forceinline__ unsigned short f2bf(float f) {
    union { float f; unsigned int i; } t; t.f = f;
    unsigned int x = t.i;
    unsigned int r = (x + 0x7fffu + ((x >> 16) & 1u)) >> 16;
    return (unsigned short)r;
}
__device__ __forceinline__ __bf16 us2bf(unsigned short u) {
    union { unsigned short u; __bf16 b; } t; t.u = u; return t.b;
}
__device__ __forceinline__ unsigned short bf2us(__bf16 v) {
    union { __bf16 b; unsigned short u; } t; t.b = v; return t.u;
}
__device__ __forceinline__ v8f v8f_zero() {
    v8f z = {0.f,0.f,0.f,0.f,0.f,0.f,0.f,0.f}; return z;
}
__device__ __forceinline__ void unpack_u4(const uint4 u, v16bf& a, const int off) {
    a[off+0] = us2bf((unsigned short)(u.x & 0xffffu)); a[off+1] = us2bf((unsigned short)(u.x >> 16));
    a[off+2] = us2bf((unsigned short)(u.y & 0xffffu)); a[off+3] = us2bf((unsigned short)(u.y >> 16));
    a[off+4] = us2bf((unsigned short)(u.z & 0xffffu)); a[off+5] = us2bf((unsigned short)(u.z >> 16));
    a[off+6] = us2bf((unsigned short)(u.w & 0xffffu)); a[off+7] = us2bf((unsigned short)(u.w >> 16));
}
// A fragment (16x32 bf16): lane L<16 -> M=L, K 0..15; lane>=16 -> M=L-16, K 16..31
__device__ __forceinline__ void load_afrag(const unsigned short* rowptr /* + k0 */, int lo, v16bf& a) {
    uint4 u0 = *(const uint4*)(rowptr + 2 * lo);
    uint4 u1 = *(const uint4*)(rowptr + 2 * lo + 8);
    unpack_u4(u0, a, 0); unpack_u4(u1, a, 8);
}
// B fragment (32x16 bf16) from row-major (padded) LDS tile: lane L<16 -> N=L, K klo..klo+15
__device__ __forceinline__ void load_bfrag_rows(const unsigned short* Bl, int klo, int stride,
                                                int ploc, v16bf& b) {
#pragma unroll
    for (int j = 0; j < 16; ++j)
        b[j] = us2bf(Bl[(klo + j) * stride + ploc]);
}
__device__ __forceinline__ v8f wmma_bf16(const v16bf a, const v16bf b, const v8f c) {
    return __builtin_amdgcn_wmma_f32_16x16x32_bf16(false, a, false, b, (short)0, c, false, false);
}

// ---------------------------------------------------------------------------
// Tensor Data Mover: async 2D tile load global->LDS (wave-issued, TENSORcnt).
// D# built per CDNA5 ISA: group0 (count=1, lds_addr,
// global_addr, type=2) + group1 (data_size=2B, pad_enable, pad_interval,
// tensor/tile dims, row stride). pad_code: interval = 2^(code+1) DWORDs,
// pad amount = 1 DWORD (row stride grows by 2 shorts).
// ---------------------------------------------------------------------------
__device__ __forceinline__ void tdm_load_2d(unsigned lds_off, const void* gptr,
                                            unsigned tile_w, unsigned tile_h,
                                            unsigned long long row_stride_elems,
                                            unsigned pad_code)
{
#if defined(__AMDGCN__)
    const unsigned long long ga = (unsigned long long)(size_t)gptr;
    v4u g0;
    g0[0] = 1u;                                            // count=1, user mode
    g0[1] = lds_off;                                       // lds_addr (bytes)
    g0[2] = (unsigned)ga;                                  // global_addr lo
    g0[3] = (unsigned)((ga >> 32) & 0x01ffffffu) | (2u << 30); // addr hi | type=2
    v8u g1;
    g1[0] = (1u << 16) | (1u << 20) | (pad_code << 22);    // data_size=2B, pad_en, interval, amount=1DW
    g1[1] = (tile_w & 0xffffu) << 16;                      // tensor_dim0 (bits 79:48)
    g1[2] = (tile_w >> 16) | ((tile_h & 0xffffu) << 16);   // tensor_dim1 (bits 111:80)
    g1[3] = (tile_h >> 16) | ((tile_w & 0xffffu) << 16);   // tile_dim0 (bits 127:112)
    g1[4] = (tile_h & 0xffffu);                            // tile_dim1 (tile_dim2=0)
    g1[5] = (unsigned)row_stride_elems;                    // tensor_dim0_stride lo
    g1[6] = (unsigned)((row_stride_elems >> 32) & 0xffffu);
    g1[7] = 0u;
    asm volatile("tensor_load_to_lds %0, %1" :: "s"(g0), "s"(g1) : "memory");
#else
    (void)lds_off; (void)gptr; (void)tile_w; (void)tile_h;
    (void)row_stride_elems; (void)pad_code;
#endif
}

// ---------------------------------------------------------------------------
// Generic WMMA conv1x1 GEMM: out[b][m][p] = sum_k W[m][k] * X[b][k][p]
// Block: 256 threads (8 waves). Tile: M=64, N=256, K=192 (6 steps of 32).
// B operand staged by double-buffered TDM (async DMA overlapped with WMMA).
// Epilogue modes:
//   0: bf16 plain                 (qkv)
//   1: bf16 + bias                (c1, c4)
//   2: f32  res1 + v              (proj -> x1)
//   3: f32  res1 + (v+bias)*csc   (c3 -> y = xm + t*beta)
//   4: f32  res1+res2+(v+bias)*csc(c5 -> out = x1 + y + t*gamma)
// Optional per-(b,k) input scale folded into the A fragment (SCA for c3).
// ---------------------------------------------------------------------------
#define SBT_ 258   // TDM-padded LDS row stride in shorts (256 + 1 pad DWORD)

template<int MODE>
__global__ __launch_bounds__(256)
void k_gemm(const unsigned short* __restrict__ Wbf,
            const unsigned short* __restrict__ Xbf,
            const float* __restrict__ bias,
            const float* __restrict__ cscale,
            const float* __restrict__ inscale,
            const float* __restrict__ res1,
            const float* __restrict__ res2,
            float* __restrict__ outf,
            unsigned short* __restrict__ outbf,
            int Mtotal)
{
    __shared__ unsigned short __attribute__((aligned(16))) Bl[2][32 * SBT_];
    const int b   = blockIdx.z;
    const int p0  = blockIdx.x * 256;
    const int m0  = blockIdx.y * 64;
    const int tid = threadIdx.x;
    const int lane = tid & 31;
    const int wave = tid >> 5;
    const int mt  = wave & 3;
    const int ntb = (wave >> 2) * 8;
    const int lo  = (lane < 16) ? 0 : 8;
    const int klo = (lane < 16) ? 0 : 16;
    const unsigned short* Xb = Xbf + (size_t)b * KIN_ * HW_;
    const unsigned short* Wrow = Wbf + (size_t)(m0 + mt * 16 + (lane & 15)) * KIN_;

    v8f acc[8];
#pragma unroll
    for (int j = 0; j < 8; ++j) acc[j] = v8f_zero();

    // prologue: wave 0 kicks off the DMA of K-slab 0
    if (tid < 32)
        tdm_load_2d((unsigned)(size_t)&Bl[0][0], Xb + p0, 256, 32, (unsigned long long)HW_, 6);

    for (int ks = 0; ks < 6; ++ks) {
        const int k0 = ks * 32;
        WAIT_TENSOR();              // wave0: slab ks landed; others: no-op
        __syncthreads();
        if (ks < 5 && tid < 32)     // overlap DMA of slab ks+1 with compute of slab ks
            tdm_load_2d((unsigned)(size_t)&Bl[(ks + 1) & 1][0],
                        Xb + (size_t)(k0 + 32) * HW_ + p0, 256, 32,
                        (unsigned long long)HW_, 6);
        if (ks < 5)
            __builtin_prefetch(Wrow + k0 + 32, 0, 1);   // global_prefetch_b8 (W row)

        v16bf a;
        load_afrag(Wrow + k0, lo, a);
        if (inscale) {              // fold per-channel SCA scale into A (w*s)*x
            const float* sc = inscale + b * KIN_ + k0;
#pragma unroll
            for (int j = 0; j < 8; ++j) {
                a[j]     = us2bf(f2bf(bf2f(bf2us(a[j]))     * sc[2 * lo + j]));
                a[8 + j] = us2bf(f2bf(bf2f(bf2us(a[8 + j])) * sc[2 * lo + 8 + j]));
            }
        }
        const unsigned short* Bcur = &Bl[ks & 1][0];
#pragma unroll
        for (int j = 0; j < 8; ++j) {
            v16bf bf;
            const int ploc = (ntb + j) * 16 + (lane & 15);
            load_bfrag_rows(Bcur, klo, SBT_, ploc, bf);
            acc[j] = wmma_bf16(a, bf, acc[j]);
        }
        __syncthreads();
    }

    // ---- epilogue: D layout VGPR vg -> M = vg (+8 for upper half-wave), lane -> N
#pragma unroll
    for (int j = 0; j < 8; ++j) {
        const int col = p0 + (ntb + j) * 16 + (lane & 15);
#pragma unroll
        for (int vg = 0; vg < 8; ++vg) {
            const int rm = m0 + mt * 16 + vg + ((lane < 16) ? 0 : 8);
            float v = acc[j][vg];
            const size_t oidx = ((size_t)b * Mtotal + rm) * HW_ + col;
            if constexpr (MODE == 0) {
                outbf[oidx] = f2bf(v);
            } else if constexpr (MODE == 1) {
                outbf[oidx] = f2bf(v + bias[rm]);
            } else if constexpr (MODE == 2) {
                outf[oidx] = res1[oidx] + v;
            } else if constexpr (MODE == 3) {
                outf[oidx] = res1[oidx] + (v + bias[rm]) * cscale[rm];
            } else {
                outf[oidx] = res1[oidx] + res2[oidx] + (v + bias[rm]) * cscale[rm];
            }
        }
    }
}

// ---------------------------------------------------------------------------
// Attention scores: per (b,h) block. S[48][48] = q·kᵀ over K=16384 (WMMA,
// 8-way wave K-split, ds_add_f32 reduction), then scale by 1/||q|| 1/||k||
// and temperature, softmax rows, write zero-padded bf16 attn[48][64].
// ---------------------------------------------------------------------------
__global__ __launch_bounds__(256)
void k_score(const unsigned short* __restrict__ qkv,
             const float* __restrict__ invn,        // [B][2][192]
             const float* __restrict__ temperature, // [4]
             unsigned short* __restrict__ attn)     // [B*4][48][64]
{
    __shared__ float S[48 * 64];
    const int bh = blockIdx.x;
    const int b = bh >> 2, h = bh & 3;
    const int tid = threadIdx.x;
    const int lane = tid & 31;
    const int wave = tid >> 5;
    for (int i = tid; i < 48 * 64; i += 256) S[i] = 0.f;
    __syncthreads();

    const unsigned short* qb = qkv + ((size_t)b * 576 + h * CH_) * HW_;
    const unsigned short* kb = qkv + ((size_t)b * 576 + 192 + h * CH_) * HW_;
    const int lo  = (lane < 16) ? 0 : 8;
    const int klo = (lane < 16) ? 0 : 16;

    v8f acc[9];
#pragma unroll
    for (int i = 0; i < 9; ++i) acc[i] = v8f_zero();

    const int kbase = wave * 2048;
    for (int kk = 0; kk < 2048; kk += 32) {
        const int k0 = kbase + kk;
        v16bf af[3], bf[3];
#pragma unroll
        for (int mt = 0; mt < 3; ++mt) {
            const int m = mt * 16 + (lane & 15);
            load_afrag(qb + (size_t)m * HW_ + k0, lo, af[mt]);
        }
#pragma unroll
        for (int nt = 0; nt < 3; ++nt) {
            const int d = nt * 16 + (lane & 15);
            const unsigned short* p = kb + (size_t)d * HW_ + k0 + klo;
            uint4 u0 = *(const uint4*)(p);
            uint4 u1 = *(const uint4*)(p + 8);
            unpack_u4(u0, bf[nt], 0); unpack_u4(u1, bf[nt], 8);
        }
#pragma unroll
        for (int mt = 0; mt < 3; ++mt)
#pragma unroll
            for (int nt = 0; nt < 3; ++nt)
                acc[mt * 3 + nt] = wmma_bf16(af[mt], bf[nt], acc[mt * 3 + nt]);
    }

#pragma unroll
    for (int mt = 0; mt < 3; ++mt)
#pragma unroll
        for (int nt = 0; nt < 3; ++nt)
#pragma unroll
            for (int vg = 0; vg < 8; ++vg) {
                const int m = mt * 16 + vg + ((lane < 16) ? 0 : 8);
                const int n = nt * 16 + (lane & 15);
                atomicAdd(&S[m * 64 + n], acc[mt * 3 + nt][vg]);   // ds_add_f32
            }
    __syncthreads();

    if (tid < 48) {
        const int m = tid;
        const float iq = invn[(b * 2 + 0) * 192 + h * CH_ + m];
        const float tm = temperature[h];
        float l[48];
        float mx = -3.4e38f;
#pragma unroll
        for (int n = 0; n < 48; ++n) {
            float ik = invn[(b * 2 + 1) * 192 + h * CH_ + n];
            l[n] = S[m * 64 + n] * iq * ik * tm;
            mx = fmaxf(mx, l[n]);
        }
        float s = 0.f;
#pragma unroll
        for (int n = 0; n < 48; ++n) { l[n] = __expf(l[n] - mx); s += l[n]; }
        const float inv = 1.f / s;
        unsigned short* dst = attn + ((size_t)bh * 48 + m) * 64;
#pragma unroll
        for (int n = 0; n < 48; ++n) dst[n] = f2bf(l[n] * inv);
#pragma unroll
        for (int n = 48; n < 64; ++n) dst[n] = 0;
    }
}

// ---------------------------------------------------------------------------
// Attention output: out[b][h*48+m][p] = sum_d attn[m][d] * v[d][p]
// V tile (48 x 128) staged by one TDM (pad interval 64 DW -> stride 130);
// K padded to 64 with zero rows. Block: 256 thr, N tile 128, 8 waves.
// ---------------------------------------------------------------------------
#define SBV_ 130

__global__ __launch_bounds__(256)
void k_attn_out(const unsigned short* __restrict__ attn,
                const unsigned short* __restrict__ qkv,
                unsigned short* __restrict__ outbf) // [B][192][HW]
{
    __shared__ unsigned short __attribute__((aligned(16))) Bl[64 * SBV_];
    const int b = blockIdx.z, h = blockIdx.y;
    const int p0 = blockIdx.x * 128;
    const int tid = threadIdx.x;
    const int lane = tid & 31;
    const int wave = tid >> 5;
    const int bh = b * 4 + h;
    const unsigned short* vb = qkv + ((size_t)b * 576 + 384 + h * CH_) * HW_;

    if (tid < 32)   // DMA v rows 0..47 into LDS (natural layout, padded rows)
        tdm_load_2d((unsigned)(size_t)&Bl[0], vb + p0, 128, 48, (unsigned long long)HW_, 5);
    for (int i = tid; i < 16 * SBV_; i += 256)   // zero K-pad rows 48..63 meanwhile
        Bl[48 * SBV_ + i] = 0;
    WAIT_TENSOR();
    __syncthreads();

    const int lo  = (lane < 16) ? 0 : 8;
    const int klo0 = (lane < 16) ? 0 : 16;
    v8f acc[3];
#pragma unroll
    for (int i = 0; i < 3; ++i) acc[i] = v8f_zero();

#pragma unroll
    for (int ks = 0; ks < 2; ++ks) {
        const int k0 = ks * 32;
#pragma unroll
        for (int mt = 0; mt < 3; ++mt) {
            const int m = mt * 16 + (lane & 15);
            v16bf a, bfr;
            load_afrag(attn + ((size_t)bh * 48 + m) * 64 + k0, lo, a);
            load_bfrag_rows(Bl, k0 + klo0, SBV_, wave * 16 + (lane & 15), bfr);
            acc[mt] = wmma_bf16(a, bfr, acc[mt]);
        }
    }

#pragma unroll
    for (int mt = 0; mt < 3; ++mt) {
        const int col = p0 + wave * 16 + (lane & 15);
#pragma unroll
        for (int vg = 0; vg < 8; ++vg) {
            const int rm = mt * 16 + vg + ((lane < 16) ? 0 : 8); // < 48
            const int ch = h * CH_ + rm;
            outbf[((size_t)b * C_ + ch) * HW_ + col] = f2bf(acc[mt][vg]);
        }
    }
}

// ---------------------------------------------------------------------------
// Elementwise / reduction kernels
// ---------------------------------------------------------------------------
__global__ void k_cvt_bf16(const float* __restrict__ in, unsigned short* __restrict__ out, int n) {
    int i = blockIdx.x * 256 + threadIdx.x;
    if (i < n) out[i] = f2bf(in[i]);
}

__global__ void k_zero_f32(float* __restrict__ p, int n) {
    int i = blockIdx.x * 256 + threadIdx.x;
    if (i < n) p[i] = 0.f;
}

// per-pixel channel LayerNorm -> bf16
__global__ __launch_bounds__(256)
void k_ln(const float* __restrict__ x, const float* __restrict__ w,
          const float* __restrict__ bb, float eps, unsigned short* __restrict__ out)
{
    const size_t pix = (size_t)blockIdx.x * 256 + threadIdx.x;
    const int b = (int)(pix >> 14);
    const int p = (int)(pix & 16383);
    const float* xb = x + (size_t)b * C_ * HW_ + p;
    float s = 0.f, ss = 0.f;
    for (int c = 0; c < C_; ++c) { float v = xb[(size_t)c * HW_]; s += v; ss += v * v; }
    const float mu = s * (1.f / C_);
    const float rs = rsqrtf(ss * (1.f / C_) - mu * mu + eps);
    unsigned short* ob = out + (size_t)b * C_ * HW_ + p;
    for (int c = 0; c < C_; ++c)
        ob[(size_t)c * HW_] = f2bf((xb[(size_t)c * HW_] - mu) * rs * w[c] + bb[c]);
}

// fused: LN2(x1) -> meta inject (gamma,beta,g) -> xm ; then LN_n1(xm) -> bf16 act
__global__ __launch_bounds__(256)
void k_meta(const float* __restrict__ x1, const float* __restrict__ w2, const float* __restrict__ b2,
            const float* __restrict__ mig, const float* __restrict__ mib, const float* __restrict__ g,
            const float* __restrict__ n1w, const float* __restrict__ n1b,
            float* __restrict__ xm, unsigned short* __restrict__ act)
{
    const size_t pix = (size_t)blockIdx.x * 256 + threadIdx.x;
    const int b = (int)(pix >> 14);
    const int p = (int)(pix & 16383);
    const float* xb = x1 + (size_t)b * C_ * HW_ + p;
    float s = 0.f, ss = 0.f;
    for (int c = 0; c < C_; ++c) { float v = xb[(size_t)c * HW_]; s += v; ss += v * v; }
    const float mu = s * (1.f / C_);
    const float rs = rsqrtf(ss * (1.f / C_) - mu * mu + 1e-5f);
    float* xmb = xm + (size_t)b * C_ * HW_ + p;
    float s2 = 0.f, ss2 = 0.f;
    for (int c = 0; c < C_; ++c) {
        float xn = (xb[(size_t)c * HW_] - mu) * rs * w2[c] + b2[c];
        float v = (xn * mig[c] + mib[c]) * g[b * C_ + c];
        xmb[(size_t)c * HW_] = v;
        s2 += v; ss2 += v * v;
    }
    const float mu2 = s2 * (1.f / C_);
    const float rs2 = rsqrtf(ss2 * (1.f / C_) - mu2 * mu2 + 1e-6f);
    unsigned short* ab = act + (size_t)b * C_ * HW_ + p;
    for (int c = 0; c < C_; ++c)
        ab[(size_t)c * HW_] = f2bf((xmb[(size_t)c * HW_] - mu2) * rs2 * n1w[c] + n1b[c]);
}

// depthwise 3x3 'SAME' on bf16 tensor [B][CH][H][W]
__global__ __launch_bounds__(256)
void k_dwconv(const unsigned short* __restrict__ in, const float* __restrict__ w9,
              const float* __restrict__ bias, unsigned short* __restrict__ out, int CHN)
{
    const int b = blockIdx.z, c = blockIdx.y;
    const int p = blockIdx.x * 256 + threadIdx.x;
    const int y = p >> 7, x = p & 127;
    const unsigned short* ib = in + ((size_t)b * CHN + c) * HW_;
    const float* wp = w9 + c * 9;
    float s = bias ? bias[c] : 0.f;
#pragma unroll
    for (int dy = -1; dy <= 1; ++dy) {
        const int yy = y + dy;
        if ((unsigned)yy >= (unsigned)H_) continue;
#pragma unroll
        for (int dx = -1; dx <= 1; ++dx) {
            const int xx = x + dx;
            if ((unsigned)xx >= (unsigned)W_) continue;
            s += bf2f(ib[yy * W_ + xx]) * wp[(dy + 1) * 3 + (dx + 1)];
        }
    }
    out[((size_t)b * CHN + c) * HW_ + p] = f2bf(s);
}

// row L2 norms of q and k -> inv norms [B][2][192]
__global__ __launch_bounds__(256)
void k_norms(const unsigned short* __restrict__ qkv, float* __restrict__ invn)
{
    __shared__ float red[256];
    const int r = blockIdx.x;                 // 0..3071
    const int b = r / 384, rem = r % 384;     // rem<192:q, else k
    const unsigned short* base = qkv + ((size_t)b * 576 + rem) * HW_;
    float ss = 0.f;
    for (int i = threadIdx.x; i < HW_; i += 256) { float v = bf2f(base[i]); ss += v * v; }
    red[threadIdx.x] = ss; __syncthreads();
    for (int s = 128; s > 0; s >>= 1) {
        if (threadIdx.x < s) red[threadIdx.x] += red[threadIdx.x + s];
        __syncthreads();
    }
    if (threadIdx.x == 0) invn[r] = 1.f / fmaxf(sqrtf(red[0]), 1e-12f);
}

// SimpleGate: out = t1*t2 (bf16); optional per-(b,c) sum accumulation
__global__ __launch_bounds__(256)
void k_gate(const unsigned short* __restrict__ tin, unsigned short* __restrict__ out,
            float* __restrict__ sums)
{
    __shared__ float red[256];
    const int b = blockIdx.z, c = blockIdx.y;
    const int p = blockIdx.x * 256 + threadIdx.x;
    const unsigned short* t1 = tin + ((size_t)b * 2 * C_ + c) * HW_;
    const unsigned short* t2 = t1 + (size_t)C_ * HW_;
    const float v = bf2f(t1[p]) * bf2f(t2[p]);
    out[((size_t)b * C_ + c) * HW_ + p] = f2bf(v);
    if (sums) {
        red[threadIdx.x] = v; __syncthreads();
        for (int s = 128; s > 0; s >>= 1) {
            if (threadIdx.x < s) red[threadIdx.x] += red[threadIdx.x + s];
            __syncthreads();
        }
        if (threadIdx.x == 0) atomicAdd(&sums[b * C_ + c], red[0]);
    }
}

// meta gate: g[b][c] = sigmoid(meta[b]·fc_w[c] + fc_b[c])
__global__ void k_gfc(const float* __restrict__ meta, const float* __restrict__ fcw,
                      const float* __restrict__ fcb, float* __restrict__ g)
{
    const int i = blockIdx.x * 256 + threadIdx.x;
    if (i >= B_ * C_) return;
    const int b = i / C_, c = i % C_;
    float a = fcb[c];
    for (int k = 0; k < 128; ++k) a += meta[b * 128 + k] * fcw[c * 128 + k];
    g[i] = 1.f / (1.f + __expf(-a));
}

// SCA: s[b][c] = sca_b[c] + sum_j sca_w[c][j] * mean_t[b][j]
__global__ void k_sca(const float* __restrict__ sums, const float* __restrict__ sw,
                      const float* __restrict__ sb, float* __restrict__ svec)
{
    const int i = blockIdx.x * 256 + threadIdx.x;
    if (i >= B_ * C_) return;
    const int b = i / C_, c = i % C_;
    float a = sb[c];
    const float inv = 1.f / (float)HW_;
    for (int j = 0; j < C_; ++j) a += sw[c * C_ + j] * (sums[b * C_ + j] * inv);
    svec[i] = a;
}

// ---------------------------------------------------------------------------
// Host orchestration
// ---------------------------------------------------------------------------
extern "C" void kernel_launch(void* const* d_in, const int* in_sizes, int n_in,
                              void* d_out, int out_size, void* d_ws, size_t ws_size,
                              hipStream_t stream)
{
    const float* x      = (const float*)d_in[0];
    const float* meta   = (const float*)d_in[1];
    const float* ln1_w  = (const float*)d_in[2];
    const float* ln1_b  = (const float*)d_in[3];
    const float* temp   = (const float*)d_in[4];
    const float* qkv_w  = (const float*)d_in[5];
    const float* dw_w   = (const float*)d_in[6];
    const float* proj_w = (const float*)d_in[7];
    const float* ln2_w  = (const float*)d_in[8];
    const float* ln2_b  = (const float*)d_in[9];
    const float* fc_w   = (const float*)d_in[10];
    const float* fc_b   = (const float*)d_in[11];
    const float* mi_g   = (const float*)d_in[12];
    const float* mi_b   = (const float*)d_in[13];
    const float* n1_w   = (const float*)d_in[14];
    const float* n1_b   = (const float*)d_in[15];
    const float* c1_w   = (const float*)d_in[16];
    const float* c1_b   = (const float*)d_in[17];
    const float* c2_w   = (const float*)d_in[18];
    const float* c2_b   = (const float*)d_in[19];
    const float* c3_w   = (const float*)d_in[20];
    const float* c3_b   = (const float*)d_in[21];
    const float* sca_w  = (const float*)d_in[22];
    const float* sca_b  = (const float*)d_in[23];
    const float* n2_w   = (const float*)d_in[24];
    const float* n2_b   = (const float*)d_in[25];
    const float* c4_w   = (const float*)d_in[26];
    const float* c4_b   = (const float*)d_in[27];
    const float* c5_w   = (const float*)d_in[28];
    const float* c5_b   = (const float*)d_in[29];
    const float* nafB   = (const float*)d_in[30];
    const float* nafG   = (const float*)d_in[31];
    float* out = (float*)d_out;

    // ---- workspace carving ----
    char* ws = (char*)d_ws;
    size_t off = 0;
    auto carve = [&](size_t bytes) { char* p = ws + off; off += (bytes + 255) & ~(size_t)255; return p; };
    unsigned short* wq   = (unsigned short*)carve((size_t)576 * 192 * 2);
    unsigned short* wpj  = (unsigned short*)carve((size_t)192 * 192 * 2);
    unsigned short* wc1  = (unsigned short*)carve((size_t)384 * 192 * 2);
    unsigned short* wc3  = (unsigned short*)carve((size_t)192 * 192 * 2);
    unsigned short* wc4  = (unsigned short*)carve((size_t)384 * 192 * 2);
    unsigned short* wc5  = (unsigned short*)carve((size_t)192 * 192 * 2);
    float*          invn = (float*)carve((size_t)B_ * 2 * C_ * 4);
    unsigned short* attn = (unsigned short*)carve((size_t)B_ * HEADS_ * 48 * 64 * 2);
    float*          gbuf = (float*)carve((size_t)B_ * C_ * 4);
    float*          sums = (float*)carve((size_t)B_ * C_ * 4);
    float*          svec = (float*)carve((size_t)B_ * C_ * 4);
    unsigned short* act  = (unsigned short*)carve((size_t)B_ * C_ * HW_ * 2);
    unsigned short* big1 = (unsigned short*)carve((size_t)B_ * 576 * HW_ * 2);
    unsigned short* big2 = (unsigned short*)carve((size_t)B_ * 576 * HW_ * 2);
    float*          xm   = (float*)carve((size_t)B_ * C_ * HW_ * 4);
    float*          ybuf = (float*)carve((size_t)B_ * C_ * HW_ * 4);
    (void)ws_size; (void)n_in; (void)in_sizes; (void)out_size;

    const dim3 blk(256);
    // weights -> bf16
    k_cvt_bf16<<<dim3((576*192+255)/256), blk, 0, stream>>>(qkv_w,  wq,  576*192);
    k_cvt_bf16<<<dim3((192*192+255)/256), blk, 0, stream>>>(proj_w, wpj, 192*192);
    k_cvt_bf16<<<dim3((384*192+255)/256), blk, 0, stream>>>(c1_w,   wc1, 384*192);
    k_cvt_bf16<<<dim3((192*192+255)/256), blk, 0, stream>>>(c3_w,   wc3, 192*192);
    k_cvt_bf16<<<dim3((384*192+255)/256), blk, 0, stream>>>(c4_w,   wc4, 384*192);
    k_cvt_bf16<<<dim3((192*192+255)/256), blk, 0, stream>>>(c5_w,   wc5, 192*192);

    // meta gate (independent)
    k_gfc<<<dim3(6), blk, 0, stream>>>(meta, fc_w, fc_b, gbuf);

    // ---- channel attention branch ----
    k_ln<<<dim3(B_*HW_/256), blk, 0, stream>>>(x, ln1_w, ln1_b, 1e-5f, act);
    k_gemm<0><<<dim3(HW_/256, 576/64, B_), blk, 0, stream>>>(wq, act, nullptr, nullptr, nullptr,
                                                             nullptr, nullptr, nullptr, big1, 576);
    k_dwconv<<<dim3(HW_/256, 576, B_), blk, 0, stream>>>(big1, dw_w, nullptr, big2, 576);
    k_norms<<<dim3(B_*2*C_), blk, 0, stream>>>(big2, invn);
    k_score<<<dim3(B_*HEADS_), blk, 0, stream>>>(big2, invn, temp, attn);
    k_attn_out<<<dim3(HW_/128, HEADS_, B_), blk, 0, stream>>>(attn, big2, act);
    // x1 = x + proj(attn_out)  -> d_out
    k_gemm<2><<<dim3(HW_/256, 192/64, B_), blk, 0, stream>>>(wpj, act, nullptr, nullptr, nullptr,
                                                             x, nullptr, out, nullptr, 192);

    // ---- meta injection + NAF first half ----
    k_meta<<<dim3(B_*HW_/256), blk, 0, stream>>>(out, ln2_w, ln2_b, mi_g, mi_b, gbuf,
                                                 n1_w, n1_b, xm, act);
    k_gemm<1><<<dim3(HW_/256, 384/64, B_), blk, 0, stream>>>(wc1, act, c1_b, nullptr, nullptr,
                                                             nullptr, nullptr, nullptr, big1, 384);
    k_dwconv<<<dim3(HW_/256, 384, B_), blk, 0, stream>>>(big1, c2_w, c2_b, big2, 384);
    k_zero_f32<<<dim3(6), blk, 0, stream>>>(sums, B_ * C_);
    k_gate<<<dim3(HW_/256, C_, B_), blk, 0, stream>>>(big2, act, sums);
    k_sca<<<dim3(6), blk, 0, stream>>>(sums, sca_w, sca_b, svec);
    // y = xm + (c3(t * s) + c3_b) * naf_beta   (SCA folded into A fragments)
    k_gemm<3><<<dim3(HW_/256, 192/64, B_), blk, 0, stream>>>(wc3, act, c3_b, nafB, svec,
                                                             xm, nullptr, ybuf, nullptr, 192);

    // ---- NAF second half ----
    k_ln<<<dim3(B_*HW_/256), blk, 0, stream>>>(ybuf, n2_w, n2_b, 1e-6f, act);
    k_gemm<1><<<dim3(HW_/256, 384/64, B_), blk, 0, stream>>>(wc4, act, c4_b, nullptr, nullptr,
                                                             nullptr, nullptr, nullptr, big1, 384);
    k_gate<<<dim3(HW_/256, C_, B_), blk, 0, stream>>>(big1, act, nullptr);
    // out = x1 + y + (c5(u) + c5_b) * naf_gamma
    k_gemm<4><<<dim3(HW_/256, 192/64, B_), blk, 0, stream>>>(wc5, act, c5_b, nafG, nullptr,
                                                             out, ybuf, out, nullptr, 192);
}
